// Encoder_34926674051563
// MI455X (gfx1250) — matrix operand order repaired and checked by reading, not compiled, not run on hardware
//
#include <hip/hip_runtime.h>
#include <hip/hip_bf16.h>
#include <cstdint>

// ---------------------------------------------------------------------------
// CDNA5 (gfx1250) implementation of the meta-GNN encoder forward pass.
// wave32, WMMA f32_16x16x32_f16 for the two batched-GEMM hotspots.
// ---------------------------------------------------------------------------

typedef __attribute__((ext_vector_type(16))) _Float16 v16h;
typedef __attribute__((ext_vector_type(8)))  float    v8f;

union Frag16 { unsigned u[8]; v16h v; };

#define DEVI __device__ __forceinline__

DEVI float sigf(float x) { return 1.f / (1.f + __expf(-x)); }

static constexpr int NN  = 1024;   // nodes
static constexpr int BB  = 8;      // batch
static constexpr int TT  = 12;     // time
static constexpr int HH  = 64;     // hidden
static constexpr int EE  = 8192;   // edges
static constexpr int BP  = 96;     // B' = B*T rows per node
static constexpr int ROW = BP * HH;      // 6144 elems per node

// =============================== GRU ======================================
// One thread per sequence (8192 seqs). Whh transposed into LDS (48KB) so each
// gate dot is 16 contiguous float4 LDS broadcasts. h lives in registers with
// static indexing; the single dynamically-indexed h[j] comes from the row we
// wrote to global last step (same-wave store->load same address is ordered).
__global__ void gru_kernel(const float* __restrict__ input,
                           const float* __restrict__ Wih,
                           const float* __restrict__ Whh,
                           const float* __restrict__ bih,
                           const float* __restrict__ bhh,
                           float* __restrict__ outf,
                           _Float16* __restrict__ outh,
                           float* __restrict__ gru_out) {
  __shared__ float WT[192 * 64];               // WT[j][k] = Whh[k][j]
  __shared__ float wi0[192], wi1[192], bi[192], bh[192];
  const int tid = threadIdx.x;                 // 64 threads
  for (int idx = tid; idx < 64 * 192; idx += 64) {
    int k = idx / 192, j = idx % 192;
    WT[j * 64 + k] = Whh[idx];
  }
  for (int j = tid; j < 192; j += 64) {
    wi0[j] = Wih[j]; wi1[j] = Wih[192 + j]; bi[j] = bih[j]; bh[j] = bhh[j];
  }
  __syncthreads();

  const int s = blockIdx.x * 64 + tid;         // sequence id, s = n*8+b
  const int n = s >> 3, b = s & 7;
  const size_t outbase = ((size_t)n * BP + b * TT) * HH;
  const float* xin = input + (size_t)s * (TT * 2);

  float h[64];
#pragma unroll
  for (int k = 0; k < 64; ++k) h[k] = 0.f;

  for (int t = 0; t < TT; ++t) {
    const float x0 = xin[t * 2], x1 = xin[t * 2 + 1];
    for (int j = 0; j < 64; ++j) {
      float ar = 0.f, az = 0.f, an = 0.f;
      const float4* r0 = (const float4*)&WT[j * 64];
      const float4* r1 = (const float4*)&WT[(j + 64) * 64];
      const float4* r2 = (const float4*)&WT[(j + 128) * 64];
#pragma unroll
      for (int k4 = 0; k4 < 16; ++k4) {
        float4 w0 = r0[k4], w1 = r1[k4], w2 = r2[k4];
        ar += h[4*k4]*w0.x + h[4*k4+1]*w0.y + h[4*k4+2]*w0.z + h[4*k4+3]*w0.w;
        az += h[4*k4]*w1.x + h[4*k4+1]*w1.y + h[4*k4+2]*w1.z + h[4*k4+3]*w1.w;
        an += h[4*k4]*w2.x + h[4*k4+1]*w2.y + h[4*k4+2]*w2.z + h[4*k4+3]*w2.w;
      }
      float gir = bi[j]       + x0 * wi0[j]       + x1 * wi1[j];
      float giz = bi[j + 64]  + x0 * wi0[j + 64]  + x1 * wi1[j + 64];
      float gin = bi[j + 128] + x0 * wi0[j + 128] + x1 * wi1[j + 128];
      float ghr = ar + bh[j], ghz = az + bh[j + 64], ghn = an + bh[j + 128];
      float r  = sigf(gir + ghr);
      float z  = sigf(giz + ghz);
      float nc = tanhf(gin + r * ghn);
      float hj = (t == 0) ? 0.f : outf[outbase + (size_t)(t - 1) * HH + j];
      float hn = (1.f - z) * nc + z * hj;
      outf[outbase + (size_t)t * HH + j] = hn;
      outh[outbase + (size_t)t * HH + j] = (_Float16)hn;
    }
    // reload updated h row (static indices) for next step
#pragma unroll
    for (int k4 = 0; k4 < 16; ++k4) {
      float4 v = *(const float4*)&outf[outbase + (size_t)t * HH + 4 * k4];
      h[4*k4] = v.x; h[4*k4+1] = v.y; h[4*k4+2] = v.z; h[4*k4+3] = v.w;
    }
  }
#pragma unroll
  for (int k = 0; k < 64; ++k) gru_out[(size_t)s * 64 + k] = h[k];
}

// ============================ CSR build ====================================
__global__ void zero_counts(int* counts) {
  int i = blockIdx.x * blockDim.x + threadIdx.x;
  if (i < NN) counts[i] = 0;
}
__global__ void count_kernel(const int* __restrict__ dst, int* counts) {
  int e = blockIdx.x * blockDim.x + threadIdx.x;
  if (e < EE) atomicAdd(&counts[dst[e]], 1);
}
__global__ void scan_kernel(const int* __restrict__ counts, int* offsets) {
  __shared__ int tmp[NN];
  int i = threadIdx.x;
  int c = counts[i];
  tmp[i] = c;
  __syncthreads();
  for (int d = 1; d < NN; d <<= 1) {
    int v = (i >= d) ? tmp[i - d] : 0;
    __syncthreads();
    tmp[i] += v;
    __syncthreads();
  }
  offsets[i] = tmp[i] - c;                     // exclusive scan
}
// Deterministic fill: rank of edge e = #earlier edges with same dst.
__global__ void fill_kernel(const int* __restrict__ dst,
                            const int* __restrict__ offsets,
                            int* __restrict__ elist) {
  int e = blockIdx.x * blockDim.x + threadIdx.x;
  if (e >= EE) return;
  int d = dst[e], r = 0;
  for (int q = 0; q < e; ++q) r += (dst[q] == d) ? 1 : 0;
  elist[offsets[d] + r] = e;
}

// ============================ edge kernel ==================================
// 1 workgroup per edge, 128 threads = 4 waves.
//  - fused 96->16->2->8192 sigmoid MLP generates w_e in LDS (transposed f16)
//  - st_e = [out[src] | out[dst]] staged f16 in LDS
//  - alpha = leaky_relu(st_e @ w_e) via 24 x v_wmma_f32_16x16x32_f16
struct GatParams {
  const float* feature; const float* dist;
  const int* src; const int* dstp;
  const float *W1, *b1, *W2, *b2, *W3, *b3;
  const _Float16* out16;
  _Float16* alpha;
};

__global__ void edge_kernel(GatParams P) {
  __shared__ unsigned stu[BP * 68];            // st_e f16, row stride 136 halves
  __shared__ _Float16 wT[64 * 136];            // wT[n][k], stride 136
  __shared__ float featE[96], h1e[16], h2e[2];

  const int e = blockIdx.x, tid = threadIdx.x;
  const int sN = P.src[e], dN = P.dstp[e];

  if (tid < 96) {
    float v;
    if (tid < 32)      v = P.feature[sN * 32 + tid];
    else if (tid < 64) v = P.feature[dN * 32 + (tid - 32)];
    else               v = P.dist[(size_t)e * 32 + (tid - 64)];
    featE[tid] = v;
  }
  const unsigned* osrc = (const unsigned*)(P.out16 + (size_t)sN * ROW);
  const unsigned* odst = (const unsigned*)(P.out16 + (size_t)dN * ROW);
  for (int lin = tid; lin < BP * 32; lin += 128) {
    int m = lin >> 5, q = lin & 31;            // q: u32 index within 64 halves
    stu[m * 68 + q]      = osrc[m * 32 + q];
    stu[m * 68 + 32 + q] = odst[m * 32 + q];
  }
  __syncthreads();
  if (tid < 16) {
    float a = P.b1[tid];
    for (int i = 0; i < 96; ++i) a += featE[i] * P.W1[i * 16 + tid];
    h1e[tid] = sigf(a);
  }
  __syncthreads();
  if (tid < 2) {
    float a = P.b2[tid];
    for (int i = 0; i < 16; ++i) a += h1e[i] * P.W2[i * 2 + tid];
    h2e[tid] = sigf(a);
  }
  __syncthreads();
  const float g0 = h2e[0], g1 = h2e[1];
  for (int q = 0; q < 64; ++q) {               // 8192 weights, coalesced
    int idx = q * 128 + tid;
    float w = sigf(P.b3[idx] + g0 * P.W3[idx] + g1 * P.W3[8192 + idx]);
    int k = idx >> 6, nn2 = idx & 63;          // w_e[k][n] -> wT[n][k]
    wT[nn2 * 136 + k] = (_Float16)w;
  }
  __syncthreads();

  const int lane = tid & 31, wid = tid >> 5;
  const int r = lane & 15, hi = lane >> 4;
  const int ntile = wid;                       // 4 waves cover n 0..63
  const unsigned* wtu = (const unsigned*)wT;

  for (int mt = 0; mt < 6; ++mt) {
    v8f acc = {};
#pragma unroll
    for (int kt = 0; kt < 4; ++kt) {
      Frag16 A, Bf;
      int kb = kt * 32 + hi * 8;
      int ab = (mt * 16 + r) * 68 + (kb >> 1);
#pragma unroll
      for (int p = 0; p < 4; ++p) { A.u[p] = stu[ab + p]; A.u[4 + p] = stu[ab + 8 + p]; }
      int nb = ((ntile * 16 + r) * 136 + kt * 32 + hi * 8) >> 1;
#pragma unroll
      for (int p = 0; p < 4; ++p) { Bf.u[p] = wtu[nb + p]; Bf.u[4 + p] = wtu[nb + 8 + p]; }
      acc = __builtin_amdgcn_wmma_f32_16x16x32_f16(false, A.v, false, Bf.v,
                                                   (short)0, acc, false, false);
    }
    size_t base = ((size_t)e * BP + mt * 16 + hi * 8) * (size_t)HH + ntile * 16 + r;
#pragma unroll
    for (int j = 0; j < 8; ++j) {
      float c = acc[j];
      P.alpha[base + (size_t)j * HH] = (_Float16)(c > 0.f ? c : 0.01f * c);
    }
  }
}

// ============================ node kernel ==================================
// 1 workgroup per node, 128 threads = 4 waves.
//  - CSR two-pass segment softmax over incoming edges (alpha is L2-resident)
//  - GAT residual:  out' = out + relu(agg * sigmoid(gat_w))
//  - fused meta-MLPs generate W_z (first half of 2H x H) and W_i2h + 3 biases
//  - state = (1 - sigmoid(out'@Wz + bz)) * tanh(out'@Wi + bi + bh)
struct Mlp { const float *W1, *b1, *W2, *b2, *W3, *b3; };
struct NodeParams {
  const float* feature; const int* src; const float* gatw;
  const int *counts, *offsets, *elist;
  const _Float16* alpha; const _Float16* out16; const float* outf;
  float* state_out;
  Mlp zw, zb, iw, ib, hb;
};

__global__ void node_kernel(NodeParams P) {
  __shared__ _Float16 st16[BP * 72];           // out' f16, stride 72
  __shared__ _Float16 BzT[64 * 72];            // Wz[n][k]
  __shared__ _Float16 BiT[64 * 72];            // Wi[n][k]
  __shared__ int earr[512], sarr[512];
  __shared__ float featL[32], h1s[5 * 16], h2s[5 * 2], sc[3];

  const int n = blockIdx.x, tid = threadIdx.x;
  const int deg = P.counts[n], off = P.offsets[n];

  if (tid < 32) featL[tid] = P.feature[n * 32 + tid];
  const int dcap = deg < 512 ? deg : 512;
  for (int q = tid; q < dcap; q += 128) {
    int e = P.elist[off + q];
    earr[q] = e; sarr[q] = P.src[e];
  }
  __syncthreads();

  // --- tiny sigmoid MLP heads: 0=zW 1=zB 2=iW 3=iB 4=hB ---
  if (tid < 80) {
    int mid = tid >> 4, j = tid & 15;
    const float *W1, *b1;
    if      (mid == 0) { W1 = P.zw.W1; b1 = P.zw.b1; }
    else if (mid == 1) { W1 = P.zb.W1; b1 = P.zb.b1; }
    else if (mid == 2) { W1 = P.iw.W1; b1 = P.iw.b1; }
    else if (mid == 3) { W1 = P.ib.W1; b1 = P.ib.b1; }
    else               { W1 = P.hb.W1; b1 = P.hb.b1; }
    float a = b1[j];
    for (int i = 0; i < 32; ++i) a += featL[i] * W1[i * 16 + j];
    h1s[mid * 16 + j] = sigf(a);
  }
  __syncthreads();
  if (tid < 10) {
    int mid = tid >> 1, j = tid & 1;
    const float *W2, *b2;
    if      (mid == 0) { W2 = P.zw.W2; b2 = P.zw.b2; }
    else if (mid == 1) { W2 = P.zb.W2; b2 = P.zb.b2; }
    else if (mid == 2) { W2 = P.iw.W2; b2 = P.iw.b2; }
    else if (mid == 3) { W2 = P.ib.W2; b2 = P.ib.b2; }
    else               { W2 = P.hb.W2; b2 = P.hb.b2; }
    float a = b2[j];
    for (int i = 0; i < 16; ++i) a += h1s[mid * 16 + i] * W2[i * 2 + j];
    h2s[mid * 2 + j] = sigf(a);
  }
  __syncthreads();
  if (tid < 3) {                               // scalar biases bz, bi, bh
    int mm = (tid == 0) ? 1 : (tid == 1) ? 3 : 4;
    const float *W3, *b3;
    if      (mm == 1) { W3 = P.zb.W3; b3 = P.zb.b3; }
    else if (mm == 3) { W3 = P.ib.W3; b3 = P.ib.b3; }
    else              { W3 = P.hb.W3; b3 = P.hb.b3; }
    sc[tid] = sigf(b3[0] + h2s[mm * 2] * W3[0] + h2s[mm * 2 + 1] * W3[1]);
  }

  // --- generate W_z (rows 0..63 of [2H,H]) and W_i2h, transposed f16 ---
  const float z0 = h2s[0], z1 = h2s[1], i0 = h2s[4], i1 = h2s[5];
  for (int q = 0; q < 32; ++q) {
    int idx = q * 128 + tid;                   // 4096 values, coalesced
    float az = P.zw.b3[idx] + z0 * P.zw.W3[idx] + z1 * P.zw.W3[8192 + idx];
    float ai = P.iw.b3[idx] + i0 * P.iw.W3[idx] + i1 * P.iw.W3[4096 + idx];
    int k = idx >> 6, nn2 = idx & 63;
    BzT[nn2 * 72 + k] = (_Float16)sigf(az);
    BiT[nn2 * 72 + k] = (_Float16)sigf(ai);
  }

  // --- segment softmax aggregation + GAT residual ---
  const float sgw = sigf(P.gatw[0]);
  for (int lin = tid; lin < ROW; lin += 128) {
    int m = lin >> 6, hcol = lin & 63;
    float outv = P.outf[(size_t)n * ROW + lin];
    float aggv = 0.f;
    if (deg > 0) {
      float amax = -1e30f;
      for (int q = 0; q < deg; ++q) {
        int e = (q < 512) ? earr[q] : P.elist[off + q];
        float a = (float)P.alpha[((size_t)e * BP + m) * HH + hcol];
        amax = fmaxf(amax, a);
      }
      float s = 0.f, num = 0.f;
      for (int q = 0; q < deg; ++q) {
        int e  = (q < 512) ? earr[q] : P.elist[off + q];
        int sv = (q < 512) ? sarr[q] : P.src[e];
        float a  = (float)P.alpha[((size_t)e * BP + m) * HH + hcol];
        float ex = __expf(a - amax);
        s += ex;
        num += ex * (float)P.out16[(size_t)sv * ROW + lin];
      }
      aggv = num / (s > 0.f ? s : 1.f);
    }
    float ns = fmaxf(aggv * sgw, 0.f);
    st16[m * 72 + hcol] = (_Float16)(outv + ns);
  }
  __syncthreads();

  // --- z / i2h GEMMs via WMMA, then state ---
  const int lane = tid & 31, wid = tid >> 5;
  const int r = lane & 15, hi = lane >> 4;
  const int ntile = wid;
  const unsigned* su = (const unsigned*)st16;
  const unsigned* bz = (const unsigned*)BzT;
  const unsigned* bi = (const unsigned*)BiT;
  const float bzv = sc[0], biv = sc[1], bhv = sc[2];

  for (int mt = 0; mt < 6; ++mt) {
    v8f cz = {}, ci = {};
#pragma unroll
    for (int kt = 0; kt < 2; ++kt) {
      Frag16 A, Bz, Bi;
      int kb = kt * 32 + hi * 8;
      int ab = (mt * 16 + r) * 36 + (kb >> 1);
#pragma unroll
      for (int p = 0; p < 4; ++p) { A.u[p] = su[ab + p]; A.u[4 + p] = su[ab + 8 + p]; }
      int nb = ((ntile * 16 + r) * 72 + kt * 32 + hi * 8) >> 1;
#pragma unroll
      for (int p = 0; p < 4; ++p) { Bz.u[p] = bz[nb + p]; Bi.u[p] = bi[nb + p]; Bz.u[4 + p] = bz[nb + 8 + p]; Bi.u[4 + p] = bi[nb + 8 + p]; }
      cz = __builtin_amdgcn_wmma_f32_16x16x32_f16(false, A.v, false, Bz.v,
                                                  (short)0, cz, false, false);
      ci = __builtin_amdgcn_wmma_f32_16x16x32_f16(false, A.v, false, Bi.v,
                                                  (short)0, ci, false, false);
    }
    size_t base = (size_t)n * ROW + (size_t)(mt * 16 + hi * 8) * HH + ntile * 16 + r;
#pragma unroll
    for (int j = 0; j < 8; ++j) {
      float z  = sigf(cz[j] + bzv);
      float tv = tanhf(ci[j] + biv + bhv);
      P.state_out[base + (size_t)j * HH] = (1.f - z) * tv;
    }
  }
}

// =============================== host ======================================
extern "C" void kernel_launch(void* const* d_in, const int* in_sizes, int n_in,
                              void* d_out, int out_size, void* d_ws, size_t ws_size,
                              hipStream_t stream) {
  (void)in_sizes; (void)n_in; (void)out_size; (void)ws_size;
  const float* input   = (const float*)d_in[0];
  const float* feature = (const float*)d_in[1];
  const int*   src     = (const int*)d_in[2];
  const int*   dstp    = (const int*)d_in[3];
  const float* dist    = (const float*)d_in[4];
  const float* gWih    = (const float*)d_in[5];
  const float* gWhh    = (const float*)d_in[6];
  const float* gbih    = (const float*)d_in[7];
  const float* gbhh    = (const float*)d_in[8];
  const float* gatW1   = (const float*)d_in[9];
  const float* gatb1   = (const float*)d_in[10];
  const float* gatW2   = (const float*)d_in[11];
  const float* gatb2   = (const float*)d_in[12];
  const float* gatW3   = (const float*)d_in[13];
  const float* gatb3   = (const float*)d_in[14];
  const float* gatw    = (const float*)d_in[15];
  auto mlp = [&](int base) {
    Mlp m;
    m.W1 = (const float*)d_in[base + 0]; m.b1 = (const float*)d_in[base + 1];
    m.W2 = (const float*)d_in[base + 2]; m.b2 = (const float*)d_in[base + 3];
    m.W3 = (const float*)d_in[base + 4]; m.b3 = (const float*)d_in[base + 5];
    return m;
  };
  Mlp zw = mlp(16), zb = mlp(22), iw = mlp(40), ib = mlp(46), hb = mlp(58);
  // md_r_* (28..39) and md_h2h weight (52..57) are dead: r multiplies prev==0.

  // workspace carve-out
  size_t o = 0;
  auto take = [&](size_t bytes) {
    o = (o + 255) & ~(size_t)255;
    size_t r = o; o += bytes; return r;
  };
  char* ws = (char*)d_ws;
  float*     outf    = (float*)(ws + take((size_t)NN * ROW * 4));       // 25.2 MB
  _Float16*  out16   = (_Float16*)(ws + take((size_t)NN * ROW * 2));    // 12.6 MB
  _Float16*  alpha   = (_Float16*)(ws + take((size_t)EE * ROW * 2));    // 100.7 MB (L2-resident)
  int*       counts  = (int*)(ws + take(NN * 4));
  int*       offsets = (int*)(ws + take(NN * 4));
  int*       elist   = (int*)(ws + take(EE * 4));

  float* gru_out   = (float*)d_out;                  // [N,B,H]
  float* state_out = (float*)d_out + (size_t)NN * BB * HH;  // [N,B*T,H]

  zero_counts<<<4, 256, 0, stream>>>(counts);
  gru_kernel<<<128, 64, 0, stream>>>(input, gWih, gWhh, gbih, gbhh,
                                     outf, out16, gru_out);
  count_kernel<<<32, 256, 0, stream>>>(dstp, counts);
  scan_kernel<<<1, 1024, 0, stream>>>(counts, offsets);
  fill_kernel<<<32, 256, 0, stream>>>(dstp, offsets, elist);

  GatParams gp;
  gp.feature = feature; gp.dist = dist; gp.src = src; gp.dstp = dstp;
  gp.W1 = gatW1; gp.b1 = gatb1; gp.W2 = gatW2; gp.b2 = gatb2;
  gp.W3 = gatW3; gp.b3 = gatb3;
  gp.out16 = out16; gp.alpha = alpha;
  edge_kernel<<<EE, 128, 0, stream>>>(gp);

  NodeParams np;
  np.feature = feature; np.src = src; np.gatw = gatw;
  np.counts = counts; np.offsets = offsets; np.elist = elist;
  np.alpha = alpha; np.out16 = out16; np.outf = outf;
  np.state_out = state_out;
  np.zw = zw; np.zb = zb; np.iw = iw; np.ib = ib; np.hb = hb;
  node_kernel<<<NN, 128, 0, stream>>>(np);
}
